// LayerConvolAll_85418309583434
// MI455X (gfx1250) — matrix operand order repaired and checked by reading, not compiled, not run on hardware
//
#include <hip/hip_runtime.h>

typedef __attribute__((ext_vector_type(16))) __bf16 v16bf;
typedef __attribute__((ext_vector_type(8)))  __bf16 v8bf;
typedef __attribute__((ext_vector_type(8)))  float  v8f;

// Exact parameter type for the async-LDS builtins (from hipcc diagnostic):
// pointer to gcc-style vector of 4 ints, global / LDS address spaces.
typedef int v4i_t __attribute__((vector_size(16)));
typedef __attribute__((address_space(1))) v4i_t glob_v4i;
typedef __attribute__((address_space(3))) v4i_t lds_v4i;

#define M_TOT 12288
#define N_TOT 12288
#define D_DIM 128
#define NT    32
// fold (1/T) * log2(e) into A so softmax runs in the exp2 domain:
// p = 2^(s - m) with s = (a.x) * log2(e)/T  ==  e^((a.x)/T - m')  (same softmax)
#define SCL2  (20.60992915555662f)
// NEG_PAD * log2(e) / T  (anything this negative underflows exp2 to 0)
#define PADV  (-8.851853e10f)

#if defined(__gfx1250__) && __has_builtin(__builtin_amdgcn_global_load_async_to_lds_b128) && __has_builtin(__builtin_amdgcn_s_wait_asynccnt)
#define USE_ASYNC 1
#else
#define USE_ASYNC 0
#endif

#if defined(__gfx1250__) && __has_builtin(__builtin_amdgcn_permlane16)
#define USE_PERMLANE 1
#else
#define USE_PERMLANE 0
#endif

// ---------------- Stage 1a: A = (xx_anchor @ W) * log2e/T  (store bf16) -----
__global__ __launch_bounds__(256) void proj_bf16_kernel(const float* __restrict__ xa,
                                                        const float* __restrict__ w,
                                                        __bf16* __restrict__ A) {
  int idx = blockIdx.x * 256 + threadIdx.x;
  int m = idx >> 7;
  int e = idx & 127;
  const float* xrow = xa + (size_t)m * D_DIM;
  float acc = 0.f;
#pragma unroll 8
  for (int k = 0; k < D_DIM; ++k) acc = fmaf(xrow[k], w[k * D_DIM + e], acc);
  A[idx] = (__bf16)(acc * SCL2);
}

// ---------------- Stage 1b: input -> bf16 ----------------
__global__ __launch_bounds__(256) void cvt_bf16_kernel(const float* __restrict__ x,
                                                       __bf16* __restrict__ y) {
  int idx = blockIdx.x * 256 + threadIdx.x;
  y[idx] = (__bf16)x[idx];
}

static __device__ __forceinline__ v16bf cat8(v8bf lo, v8bf hi) {
  return __builtin_shufflevector(lo, hi, 0, 1, 2, 3, 4, 5, 6, 7,
                                         8, 9, 10, 11, 12, 13, 14, 15);
}

// LDS store->load within a wave is in-order (DScnt), but the compiler cannot
// see cross-lane deps through LDS: pin ordering.
static __device__ __forceinline__ void wave_sync() {
  asm volatile("" ::: "memory");
  __builtin_amdgcn_wave_barrier();
}

// ---- 16-lane butterfly reductions via v_permlane16_b32 (VALU, no LDS) ------
// Nibble j of sel0 = source lane for lane j (row-local), sel1 for lanes 8..15;
// lanes 16..31 reuse the same pattern within their row => per-half reduction.
#if USE_PERMLANE
static __device__ __forceinline__ float plane16(float v, unsigned s0, unsigned s1) {
  unsigned u = __float_as_uint(v);
  u = __builtin_amdgcn_permlane16(u, u, s0, s1, false, false);
  return __uint_as_float(u);
}
static __device__ __forceinline__ float redmax16(float t) {
  t = fmaxf(t, plane16(t, 0x67452301u, 0xEFCDAB89u));  // xor 1
  t = fmaxf(t, plane16(t, 0x54761032u, 0xDCFE98BAu));  // xor 2
  t = fmaxf(t, plane16(t, 0x32107654u, 0xBA98FEDCu));  // xor 4
  t = fmaxf(t, plane16(t, 0xFEDCBA98u, 0x76543210u));  // xor 8
  return t;
}
static __device__ __forceinline__ float redsum16(float t) {
  t += plane16(t, 0x67452301u, 0xEFCDAB89u);
  t += plane16(t, 0x54761032u, 0xDCFE98BAu);
  t += plane16(t, 0x32107654u, 0xBA98FEDCu);
  t += plane16(t, 0xFEDCBA98u, 0x76543210u);
  return t;
}
#else
static __device__ __forceinline__ float redmax16(float t) {
  for (int off = 1; off < 16; off <<= 1) t = fmaxf(t, __shfl_xor(t, off, 32));
  return t;
}
static __device__ __forceinline__ float redsum16(float t) {
  for (int off = 1; off < 16; off <<= 1) t += __shfl_xor(t, off, 32);
  return t;
}
#endif

#if USE_ASYNC
// Double-buffered async copy of a 64x32 int adjacency tile into LDS.
// 128 threads x 4 async b128 copies = 8 KB. Rows padded to 36 ints so the row
// stride (144B) stays 16B-aligned and row+8 lands 32 banks away.
static __device__ __forceinline__ void issue_adj_copy(const int* __restrict__ adj,
                                                      int (*dst)[36],
                                                      int m0blk, int nbase, int tid) {
#pragma unroll
  for (int q = 0; q < 4; ++q) {
    const int chunk = q * 128 + tid;     // 512 chunks of 16B
    const int row   = chunk >> 3;        // 0..63
    const int c4    = (chunk & 7) * 4;   // int index 0..28
    const int* g = adj + (size_t)(m0blk + row) * N_TOT + nbase + c4;
    __builtin_amdgcn_global_load_async_to_lds_b128(
        (glob_v4i*)g, (lds_v4i*)&dst[row][c4], 0, 0);
  }
}
#endif

// ---------------- Stage 2: fused masked-softmax attention -------------------
// Block = 128 threads = 4 wave32s; each wave owns 16 anchor rows.
// N streamed in tiles of 32 with online softmax (exp2 domain); adjacency
// sparsity skips score-WMMA/exp work for fully-masked 16x16 subtiles and
// whole 16x32 tiles (wave-uniform ballot branches).
__global__ __launch_bounds__(128) void attn_fused_kernel(const __bf16* __restrict__ Abf,
                                                         const __bf16* __restrict__ Xbf,
                                                         const int*    __restrict__ adj,
                                                         float*        __restrict__ out) {
  __shared__ __bf16 VT[128][40];     // transposed input tile  (rows 80B)
  __shared__ __bf16 Pw[4][16][40];   // per-wave P staging     (rows 80B)
#if USE_ASYNC
  __shared__ int Aadj[2][64][36];    // double-buffered adj tile
#endif

  const int tid   = threadIdx.x;
  const int wave  = tid >> 5;
  const int lane  = tid & 31;
  const int half  = lane >> 4;
  const int l15   = lane & 15;
  const int m0blk = blockIdx.x * 64;
  const int m0    = m0blk + wave * 16;

  // Resident A operand: 16x128 bf16 in WMMA A-layout (per ISA 16-bit A 16x32:
  // lane-half 0 holds K=k0+0..7,k0+16..23; half 1 holds K=k0+8..15,k0+24..31).
  v16bf a[4];
  {
    const __bf16* arow = Abf + (size_t)(m0 + l15) * D_DIM;
#pragma unroll
    for (int kk = 0; kk < 4; ++kk) {
      const int k0 = kk * 32;
      v8bf lo = *(const v8bf*)(arow + k0 + half * 8);
      v8bf hi = *(const v8bf*)(arow + k0 + 16 + half * 8);
      a[kk] = cat8(lo, hi);
    }
  }

  v8f O[8];                 // 16x128 f32 accumulator in C/D layout
#pragma unroll
  for (int c = 0; c < 8; ++c) O[c] = v8f{0, 0, 0, 0, 0, 0, 0, 0};
  float rmax[8], rsum[8];   // stats for row m = r + 8*half (replicated per half)
#pragma unroll
  for (int r = 0; r < 8; ++r) { rmax[r] = PADV; rsum[r] = 0.f; }

#if !USE_ASYNC
  const int* abase = adj + (size_t)(m0 + 8 * half) * N_TOT + l15;
#endif

#if USE_ASYNC
  issue_adj_copy(adj, Aadj[0], m0blk, 0, tid);
#endif

  for (int it = 0; it < N_TOT / NT; ++it) {
    const int nb = it * NT;
#if USE_ASYNC
    const int buf = it & 1;
#endif

    // ---- cooperative transpose-stage of the 32x128 input tile into VT ----
    {
      const int row = tid >> 2;          // 0..31
      const int d0  = (tid & 3) * 32;    // 0,32,64,96
      const __bf16* src = Xbf + (size_t)(nb + row) * D_DIM + d0;
#pragma unroll
      for (int q = 0; q < 4; ++q) {
        v8bf x = *(const v8bf*)(src + q * 8);
#pragma unroll
        for (int i = 0; i < 8; ++i) VT[d0 + q * 8 + i][row] = x[i];
      }
    }

#if USE_ASYNC
    // prefetch next adj tile, then wait for the current one (in-order async)
    if (nb + NT < N_TOT) {
      issue_adj_copy(adj, Aadj[buf ^ 1], m0blk, nb + NT, tid);
      __builtin_amdgcn_s_wait_asynccnt(4);
    } else {
      __builtin_amdgcn_s_wait_asynccnt(0);
    }
#endif
    __syncthreads();

    // ---- adjacency masks: bit r of msk[sub] = adj[m0+r+8h][nb+16sub+l15]>0 --
    unsigned msk[2] = {0u, 0u};
#pragma unroll
    for (int sub = 0; sub < 2; ++sub) {
#pragma unroll
      for (int r = 0; r < 8; ++r) {
#if USE_ASYNC
        const int v = Aadj[buf][wave * 16 + 8 * half + r][sub * 16 + l15];
#else
        const int v = abase[(size_t)r * N_TOT + nb + sub * 16];
#endif
        msk[sub] |= (v > 0) ? (1u << r) : 0u;
      }
    }
    int live[2];
#pragma unroll
    for (int sub = 0; sub < 2; ++sub) live[sub] = (__ballot(msk[sub] != 0u) != 0ull);

    if (live[0] | live[1]) {
      // ---- scores for live subtiles: S = (A * log2e/T) @ X^T ----
      v8f S[2];
#pragma unroll
      for (int sub = 0; sub < 2; ++sub) {
        if (live[sub]) {
          const __bf16* xrow = Xbf + (size_t)(nb + sub * 16 + l15) * D_DIM + half * 8;
          S[sub] = v8f{0, 0, 0, 0, 0, 0, 0, 0};
#pragma unroll
          for (int kk = 0; kk < 4; ++kk) {
            v8bf b0 = *(const v8bf*)(xrow + kk * 32);
            v8bf b1 = *(const v8bf*)(xrow + kk * 32 + 16);
            S[sub] = __builtin_amdgcn_wmma_f32_16x16x32_bf16(false, a[kk], false, cat8(b0, b1),
                                                             (short)0, S[sub], false, false);
          }
        }
      }

      // ---- per-row max over live subtiles, then half-wave reduction ----
      float tmv[8];
#pragma unroll
      for (int r = 0; r < 8; ++r) tmv[r] = PADV;
#pragma unroll
      for (int sub = 0; sub < 2; ++sub) {
        if (live[sub]) {
#pragma unroll
          for (int r = 0; r < 8; ++r) {
            const float s = ((msk[sub] >> r) & 1u) ? S[sub][r] : PADV;
            tmv[r] = fmaxf(tmv[r], s);
          }
        }
      }
      float nm[8];
      bool chg = false;
#pragma unroll
      for (int r = 0; r < 8; ++r) {
        nm[r] = fmaxf(rmax[r], redmax16(tmv[r]));
        chg = chg || (nm[r] > rmax[r]);
      }

      // ---- rescale accumulator only when some row max actually moved ----
      if (__ballot(chg) != 0ull) {
#pragma unroll
        for (int r = 0; r < 8; ++r) {
          const float corr = exp2f(rmax[r] - nm[r]);
          rsum[r] *= corr;
          rmax[r] = nm[r];
#pragma unroll
          for (int c2 = 0; c2 < 8; ++c2) O[c2][r] *= corr;
        }
      }

      // ---- exp2 + stage P into LDS (zeros for dead subtiles) ----
      float ps[8];
#pragma unroll
      for (int r = 0; r < 8; ++r) ps[r] = 0.f;
#pragma unroll
      for (int sub = 0; sub < 2; ++sub) {
        if (live[sub]) {
#pragma unroll
          for (int r = 0; r < 8; ++r) {
            const float s = ((msk[sub] >> r) & 1u) ? S[sub][r] : PADV;
            const float p = exp2f(s - rmax[r]);
            ps[r] += p;
            Pw[wave][r + 8 * half][sub * 16 + l15] = (__bf16)p;
          }
        } else {
#pragma unroll
          for (int r = 0; r < 8; ++r)
            Pw[wave][r + 8 * half][sub * 16 + l15] = (__bf16)0.f;
        }
      }
#pragma unroll
      for (int r = 0; r < 8; ++r) rsum[r] += redsum16(ps[r]);

      wave_sync();

      // ---- O += P(16x32) @ V(32x128) ----
      {
        const __bf16* prow = &Pw[wave][l15][0];
        v8bf plo = *(const v8bf*)(prow + 8 * half);
        v8bf phi = *(const v8bf*)(prow + 16 + 8 * half);
        v16bf pA = cat8(plo, phi);
#pragma unroll
        for (int c2 = 0; c2 < 8; ++c2) {
          const __bf16* vrow = &VT[c2 * 16 + l15][half * 8];
          v8bf b0 = *(const v8bf*)(vrow);
          v8bf b1 = *(const v8bf*)(vrow + 16);
          O[c2] = __builtin_amdgcn_wmma_f32_16x16x32_bf16(false, pA, false, cat8(b0, b1),
                                                          (short)0, O[c2], false, false);
        }
      }
      wave_sync();
    }
    __syncthreads();   // VT / Pw / Aadj[buf] reused next iteration
  }

  // ---- normalize and store: out[m, d] = O / rowsum ----
  float invs[8];
#pragma unroll
  for (int r = 0; r < 8; ++r) invs[r] = 1.f / rsum[r];
#pragma unroll
  for (int c2 = 0; c2 < 8; ++c2) {
#pragma unroll
    for (int r = 0; r < 8; ++r) {
      out[(size_t)(m0 + r + 8 * half) * D_DIM + c2 * 16 + l15] = O[c2][r] * invs[r];
    }
  }
}

extern "C" void kernel_launch(void* const* d_in, const int* in_sizes, int n_in,
                              void* d_out, int out_size, void* d_ws, size_t ws_size,
                              hipStream_t stream) {
  (void)in_sizes; (void)n_in; (void)out_size; (void)ws_size;
  const float* xx  = (const float*)d_in[0];   // [M, D]
  const float* inp = (const float*)d_in[1];   // [N, D]
  const int*   adj = (const int*)d_in[2];     // [M, N]
  const float* w   = (const float*)d_in[3];   // [D, D]
  float* out = (float*)d_out;                 // [M, D]

  __bf16* Abf = (__bf16*)d_ws;                    // M*D bf16 (pre-scaled by log2e/T)
  __bf16* Xbf = Abf + (size_t)M_TOT * D_DIM;      // N*D bf16

  const int elems = M_TOT * D_DIM;
  proj_bf16_kernel<<<elems / 256, 256, 0, stream>>>(xx, w, Abf);
  cvt_bf16_kernel<<<(N_TOT * D_DIM) / 256, 256, 0, stream>>>(inp, Xbf);
  attn_fused_kernel<<<M_TOT / 64, 128, 0, stream>>>(Abf, Xbf, adj, out);
}
